// LocalAttention_mm_7851200217267
// MI455X (gfx1250) — matrix operand order repaired and checked by reading, not compiled, not run on hardware
//
#include <hip/hip_runtime.h>
#include <hip/hip_bf16.h>

// ---------------------------------------------------------------------------
// Fused Swin-window attention for MI455X (gfx1250, wave32, WMMA).
// One workgroup (256 threads = 8 waves) per 8x8 window; 2048 windows.
// All GEMMs use v_wmma_f32_16x16x32_f16 with f32 accumulation.
// rel_idx / rpb tables staged to LDS with GLOBAL_LOAD_ASYNC_TO_LDS_B128,
// drained with s_wait_asynccnt before the attention phase.
// ---------------------------------------------------------------------------

typedef _Float16 v16h __attribute__((ext_vector_type(16)));
typedef _Float16 v8h  __attribute__((ext_vector_type(8)));
typedef float    v8f  __attribute__((ext_vector_type(8)));
typedef int      v4i  __attribute__((vector_size(16)));

#define CDIM    512
#define NHEADS  16
#define HD      32
#define NTOK    64
#define WINSZ   8
#define IMG     128
#define NWIN    2048      // 8 * 16 * 16 windows
#define PITCH   520       // halves per row for 64x512 LDS panels (16B-aligned rows)
#define PP      72        // halves per row for per-wave P panel (144B rows, 16B aligned)

// LDS offsets in halves
#define OFF_X   0                         // X stage (64x520) / P area (8 waves x 64x72)
#define REGION0 36864                     // max(64*520, 8*64*72)
#define OFF_Q   (REGION0)                 // Q panel, later O panel (64x520)
#define OFF_K   (OFF_Q + NTOK * PITCH)
#define OFF_V   (OFF_K + NTOK * PITCH)
#define PANEL_HALVES (OFF_V + NTOK * PITCH)        // 136704 halves = 273408 bytes
// Byte offsets for the async-staged tables (16B aligned)
#define TAB_REL_BYTE (PANEL_HALVES * 2)            // 4096 int32  = 16384 B
#define TAB_RPB_BYTE (TAB_REL_BYTE + 16384)        // 3600 float  = 14400 B
#define SMEM_BYTES   (TAB_RPB_BYTE + 14400)        // 304192 B < 320 KB

#define AS1 __attribute__((address_space(1)))
#define AS3 __attribute__((address_space(3)))

#if __has_builtin(__builtin_amdgcn_global_load_async_to_lds_b128) && \
    __has_builtin(__builtin_amdgcn_s_wait_asynccnt)
#define HAVE_ASYNC_LDS 1
#else
#define HAVE_ASYNC_LDS 0
#endif

#if HAVE_ASYNC_LDS
// 16-byte async copy global -> LDS (tracked by ASYNCcnt).
// Builtin signature (from hipcc diagnostic): param 0 is `v4i __device__*`
// (AS1 int4 pointer); param 1 is the LDS-side int4 pointer.
static __device__ inline void async_copy16(const void* g, void* l) {
  __builtin_amdgcn_global_load_async_to_lds_b128(
      (AS1 v4i*)(uintptr_t)g,
      (AS3 v4i*)(unsigned)(uintptr_t)l,
      /*imm offset=*/0, /*cpol=*/0);
}
#endif

static __device__ inline v8f wmma_f16(v16h a, v16h b, v8f c) {
  // D = A(16x32 f16) * B(32x16 f16) + C(16x16 f32)
  return __builtin_amdgcn_wmma_f32_16x16x32_f16(
      /*neg_a=*/false, a, /*neg_b=*/false, b,
      /*c_mod=*/(short)0, c, /*reuse_a=*/false, /*reuse_b=*/false);
}

static __device__ inline v8f wmma_f16_ra(v16h a, v16h b, v8f c) {
  // Same, with A-operand reuse hint (previous WMMA is identical and shares A).
  return __builtin_amdgcn_wmma_f32_16x16x32_f16(
      false, a, false, b, (short)0, c, /*reuse_a=*/true, /*reuse_b=*/false);
}

// A-matrix fragment (16x32 f16) from a row-major LDS panel.
// A layout: lane m = lane&15; elems 0..15 hold K = k0 + hi*16 + 0..15  (hi = lane>=16).
static __device__ inline v16h frag_a_row(const _Float16* __restrict__ base,
                                         int row0, int k0, int pitch, int lane) {
  const int r  = row0 + (lane & 15);
  const int hi = lane >> 4;
  const _Float16* p = base + (size_t)r * pitch + k0 + hi * 16;
  v8h lo = *(const v8h*)(p);
  v8h hh = *(const v8h*)(p + 8);
  v16h out;
#pragma unroll
  for (int e = 0; e < 8; ++e) { out[e] = lo[e]; out[e + 8] = hh[e]; }
  return out;
}

// B-matrix fragment (32x16 f16) where B(k,n) = base[row0+n][k0+k] (row-major source).
// B layout (contiguous k, same as A here): lane n = lane&15; elems 0..15 hold K = k0 + hi*16 + 0..15.
static __device__ inline v16h frag_b_row(const _Float16* __restrict__ base,
                                         int row0, int k0, int pitch, int lane) {
  const int n  = row0 + (lane & 15);
  const int hi = lane >> 4;
  const _Float16* p = base + (size_t)n * pitch + k0 + hi * 16;
  v8h lo = *(const v8h*)(p);
  v8h hh = *(const v8h*)(p + 8);
  v16h out;
#pragma unroll
  for (int e = 0; e < 8; ++e) { out[e] = lo[e]; out[e + 8] = hh[e]; }
  return out;
}

// B fragment for W^T GEMMs: B(k,n) = W[n0+n][k0+k], W row-major f32 (ld = ldw).
// Contiguous 16-float run per lane, converted to f16.
static __device__ inline v16h frag_b_wT(const float* __restrict__ W, int ldw,
                                        int n0, int k0, int lane) {
  const float* p = W + (size_t)(n0 + (lane & 15)) * ldw + k0 + (lane >> 4) * 16;
  float4 a0 = *(const float4*)(p + 0);
  float4 a1 = *(const float4*)(p + 4);
  float4 a2 = *(const float4*)(p + 8);
  float4 a3 = *(const float4*)(p + 12);
  v16h out;
  out[0]  = (_Float16)a0.x; out[1]  = (_Float16)a0.y; out[2]  = (_Float16)a0.z; out[3]  = (_Float16)a0.w;
  out[4]  = (_Float16)a1.x; out[5]  = (_Float16)a1.y; out[6]  = (_Float16)a1.z; out[7]  = (_Float16)a1.w;
  out[8]  = (_Float16)a2.x; out[9]  = (_Float16)a2.y; out[10] = (_Float16)a2.z; out[11] = (_Float16)a2.w;
  out[12] = (_Float16)a3.x; out[13] = (_Float16)a3.y; out[14] = (_Float16)a3.z; out[15] = (_Float16)a3.w;
  return out;
}

// B fragment with B(k,n) = base[k0+k][n0+n] (column walk, strided LDS reads) — for P*V.
static __device__ inline v16h frag_b_col(const _Float16* __restrict__ base,
                                         int k0, int n0, int pitch, int lane) {
  const int n   = n0 + (lane & 15);
  const int kb  = k0 + (lane >> 4) * 16;
  v16h out;
#pragma unroll
  for (int e = 0; e < 16; ++e) out[e] = base[(size_t)(kb + e) * pitch + n];
  return out;
}

// Store f32 C/D tile to LDS as f16: value = (acc + addv) * mul.
// C layout: lane n = n0 + (lane&15); reg r holds m = m0 + r + 8*hi.
static __device__ inline void store_c_f16(_Float16* __restrict__ dst, int m0, int n0,
                                          int pitch, int lane, v8f acc,
                                          float addv, float mul) {
  const int n  = n0 + (lane & 15);
  const int hi = lane >> 4;
#pragma unroll
  for (int r = 0; r < 8; ++r)
    dst[(size_t)(m0 + r + hi * 8) * pitch + n] = (_Float16)((acc[r] + addv) * mul);
}

// Stage one 64-token window of a NCHW f32 tensor into LDS as f16 [tok][chan].
static __device__ inline void stage_window(const float* __restrict__ src,
                                           int b, int wy, int wx,
                                           _Float16* __restrict__ sX, int tid) {
  // 64 tokens x 512 channels, 4 consecutive x-pixels per iteration (float4).
  for (int i = tid; i < (CDIM * NTOK) / 4; i += 256) {
    const int lin = i * 4;
    const int c   = lin >> 6;        // channel
    const int rem = lin & 63;
    const int ty  = rem >> 3;
    const int tx4 = rem & 7;         // 0 or 4
    const int y = wy * WINSZ + ty;
    const int x = wx * WINSZ + tx4;
    const float4 f = *(const float4*)(src + (((size_t)b * CDIM + c) * IMG + y) * IMG + x);
    const int t = ty * WINSZ + tx4;
    sX[(size_t)(t + 0) * PITCH + c] = (_Float16)f.x;
    sX[(size_t)(t + 1) * PITCH + c] = (_Float16)f.y;
    sX[(size_t)(t + 2) * PITCH + c] = (_Float16)f.z;
    sX[(size_t)(t + 3) * PITCH + c] = (_Float16)f.w;
  }
}

__global__ __launch_bounds__(256)
void swin_local_attn_kernel(const float* __restrict__ xa,
                            const float* __restrict__ xb,
                            const float* __restrict__ Wq,
                            const float* __restrict__ bq,
                            const float* __restrict__ Wkv,
                            const float* __restrict__ bkv,
                            const float* __restrict__ Wo,
                            const float* __restrict__ bo,
                            const float* __restrict__ rpb,
                            const int*   __restrict__ rel_idx,
                            float* __restrict__ out) {
  extern __shared__ _Float16 smem[];
  _Float16* sX = smem + OFF_X;   // X stage, later per-wave P panels
  _Float16* sQ = smem + OFF_Q;   // Q panel, later O panel
  _Float16* sK = smem + OFF_K;
  _Float16* sV = smem + OFF_V;
  int*   ldsRel = (int*)((char*)smem + TAB_REL_BYTE);
  float* ldsRpb = (float*)((char*)smem + TAB_RPB_BYTE);

  const int tid    = threadIdx.x;
  const int wave   = tid >> 5;
  const int lane   = tid & 31;
  const int lane15 = lane & 15;
  const int hi     = lane >> 4;

  const int wi   = blockIdx.x;       // window index = b*256 + wy*16 + wx
  const int bIdx = wi >> 8;
  const int wy   = (wi >> 4) & 15;
  const int wx   = wi & 15;

  const float scale = 0.17677669529663687f;  // 32^-0.5

  // ---- Kick off async staging of bias tables into LDS (ASYNCcnt).
  //      Overlaps with phases 1-4; drained before the attention phase. ----
#if HAVE_ASYNC_LDS
  for (int i = tid; i < 1024; i += 256)                 // 4096 int32 = 1024 x 16B
    async_copy16(rel_idx + i * 4, ldsRel + i * 4);
  for (int i = tid; i < 900; i += 256)                  // 3600 f32  =  900 x 16B
    async_copy16(rpb + i * 4, ldsRpb + i * 4);
#else
  for (int i = tid; i < 4096; i += 256) ldsRel[i] = rel_idx[i];
  for (int i = tid; i < 3600; i += 256) ldsRpb[i] = rpb[i];
#endif

  // Warm L2 for weight panels (global_prefetch_b8).
#pragma unroll
  for (int j = 0; j < 4; ++j) {
    __builtin_prefetch(Wq  + tid * 256 + j * 65536, 0, 0);
    __builtin_prefetch(Wkv + tid * 512 + j * 131072, 0, 0);
    __builtin_prefetch(Wo  + tid * 256 + j * 65536, 0, 0);
  }

  // ---- Phase 1: stage Xa ----
  stage_window(xa, bIdx, wy, wx, sX, tid);
  __syncthreads();

  // ---- Phase 2: Q = (Xa * Wq^T + bq) * scale  -> sQ (f16) ----
  for (int tile = wave; tile < 128; tile += 8) {
    const int m0 = (tile >> 5) * 16;
    const int n0 = (tile & 31) * 16;
    v8f acc = {};
    for (int k0 = 0; k0 < CDIM; k0 += 32) {
      v16h a = frag_a_row(sX, m0, k0, PITCH, lane);
      v16h b = frag_b_wT(Wq, CDIM, n0, k0, lane);
      acc = wmma_f16(a, b, acc);
    }
    const float bv = bq[n0 + lane15];
    store_c_f16(sQ, m0, n0, PITCH, lane, acc, bv, scale);
  }
  __syncthreads();

  // ---- Phase 3: stage Xb (overwrite sX) ----
  stage_window(xb, bIdx, wy, wx, sX, tid);
  __syncthreads();

  // ---- Phase 4: [K|V] = Xb * Wkv^T + bkv  -> sK, sV (f16) ----
  for (int tile = wave; tile < 256; tile += 8) {
    const int m0 = (tile >> 6) * 16;
    const int nt = tile & 63;            // 0..63 over the 1024 output cols
    const int o0 = nt * 16;              // row offset into Wkv / bkv
    v8f acc = {};
    for (int k0 = 0; k0 < CDIM; k0 += 32) {
      v16h a = frag_a_row(sX, m0, k0, PITCH, lane);
      v16h b = frag_b_wT(Wkv, CDIM, o0, k0, lane);
      acc = wmma_f16(a, b, acc);
    }
    const float bv = bkv[o0 + lane15];
    _Float16* dst = (nt < 32) ? sK : sV;
    const int c0  = (nt < 32) ? o0 : (o0 - CDIM);
    store_c_f16(dst, m0, c0, PITCH, lane, acc, bv, 1.0f);
  }
#if HAVE_ASYNC_LDS
  __builtin_amdgcn_s_wait_asynccnt(0);   // table copies complete (own wave)
#endif
  __syncthreads();                       // ... and visible to all waves

  // ---- Phase 5: attention (2 heads per wave) ----
  _Float16* Pw = sX + (size_t)wave * (NTOK * PP);   // per-wave P panel

#pragma unroll 1
  for (int hh = 0; hh < 2; ++hh) {
    const int h  = wave * 2 + hh;
    const int qc = h * HD;               // column offset of this head in Q/K/V/O

    // Scores + bias + softmax + P (per 16-row block)
#pragma unroll 1
    for (int mt = 0; mt < 4; ++mt) {
      v8f s[4] = {v8f{}, v8f{}, v8f{}, v8f{}};
      v16h aq = frag_a_row(sQ, mt * 16, qc, PITCH, lane);
      // S = Q * K^T : B(c,n) = K[n][qc+c] -> row-major B fragment from sK.
      // Back-to-back identical WMMAs sharing A: set reuse_a on calls 2..4.
      {
        v16h bk0 = frag_b_row(sK,  0, qc, PITCH, lane);
        v16h bk1 = frag_b_row(sK, 16, qc, PITCH, lane);
        v16h bk2 = frag_b_row(sK, 32, qc, PITCH, lane);
        v16h bk3 = frag_b_row(sK, 48, qc, PITCH, lane);
        s[0] = wmma_f16   (aq, bk0, s[0]);
        s[1] = wmma_f16_ra(aq, bk1, s[1]);
        s[2] = wmma_f16_ra(aq, bk2, s[2]);
        s[3] = wmma_f16_ra(aq, bk3, s[3]);
      }
      // relative position bias (LDS-resident tables)
#pragma unroll
      for (int nt = 0; nt < 4; ++nt) {
        const int n = nt * 16 + lane15;
#pragma unroll
        for (int r = 0; r < 8; ++r) {
          const int m  = mt * 16 + r + hi * 8;
          const int ri = ldsRel[m * NTOK + n];
          s[nt][r] += ldsRpb[ri * NHEADS + h];
        }
      }
      // row softmax: each 64-wide row lives in 4 accumulators x 16 lanes
#pragma unroll
      for (int r = 0; r < 8; ++r) {
        float mx = fmaxf(fmaxf(s[0][r], s[1][r]), fmaxf(s[2][r], s[3][r]));
        mx = fmaxf(mx, __shfl_xor(mx, 1, 32));
        mx = fmaxf(mx, __shfl_xor(mx, 2, 32));
        mx = fmaxf(mx, __shfl_xor(mx, 4, 32));
        mx = fmaxf(mx, __shfl_xor(mx, 8, 32));
        float e0 = __expf(s[0][r] - mx);
        float e1 = __expf(s[1][r] - mx);
        float e2 = __expf(s[2][r] - mx);
        float e3 = __expf(s[3][r] - mx);
        float sum = e0 + e1 + e2 + e3;
        sum += __shfl_xor(sum, 1, 32);
        sum += __shfl_xor(sum, 2, 32);
        sum += __shfl_xor(sum, 4, 32);
        sum += __shfl_xor(sum, 8, 32);
        const float inv = 1.0f / sum;
        const int m = mt * 16 + r + hi * 8;
        Pw[(size_t)m * PP +  0 + lane15] = (_Float16)(e0 * inv);
        Pw[(size_t)m * PP + 16 + lane15] = (_Float16)(e1 * inv);
        Pw[(size_t)m * PP + 32 + lane15] = (_Float16)(e2 * inv);
        Pw[(size_t)m * PP + 48 + lane15] = (_Float16)(e3 * inv);
      }
    }

    // O_h = P * V_h  (K = 64 -> two chained WMMAs); O overwrites sQ columns of head h
#pragma unroll 1
    for (int mt = 0; mt < 4; ++mt) {
#pragma unroll
      for (int cv = 0; cv < 2; ++cv) {
        v8f acc = {};
#pragma unroll
        for (int kk = 0; kk < 2; ++kk) {
          v16h a  = frag_a_row(Pw, mt * 16, kk * 32, PP, lane);
          v16h bv = frag_b_col(sV, kk * 32, qc + cv * 16, PITCH, lane);
          acc = wmma_f16(a, bv, acc);
        }
        store_c_f16(sQ, mt * 16, qc + cv * 16, PITCH, lane, acc, 0.0f, 1.0f);
      }
    }
  }
  __syncthreads();

  // ---- Phase 6: Y = O * Wo^T + bo, scatter to NCHW f32 ----
  const _Float16* sO = sQ;
  for (int tile = wave; tile < 128; tile += 8) {
    const int m0 = (tile >> 5) * 16;
    const int n0 = (tile & 31) * 16;
    v8f acc = {};
    for (int k0 = 0; k0 < CDIM; k0 += 32) {
      v16h a = frag_a_row(sO, m0, k0, PITCH, lane);
      v16h b = frag_b_wT(Wo, CDIM, n0, k0, lane);
      acc = wmma_f16(a, b, acc);
    }
    const int   nch = n0 + lane15;
    const float bv  = bo[nch];
#pragma unroll
    for (int r = 0; r < 8; ++r) {
      const int m = m0 + r + hi * 8;
      const int y = wy * WINSZ + (m >> 3);
      const int x = wx * WINSZ + (m & 7);
      out[(((size_t)bIdx * CDIM + nch) * IMG + y) * IMG + x] = acc[r] + bv;
    }
  }
}

extern "C" void kernel_launch(void* const* d_in, const int* in_sizes, int n_in,
                              void* d_out, int out_size, void* d_ws, size_t ws_size,
                              hipStream_t stream) {
  (void)in_sizes; (void)n_in; (void)out_size; (void)d_ws; (void)ws_size;
  const float* xa      = (const float*)d_in[0];
  const float* xb      = (const float*)d_in[1];
  const float* Wq      = (const float*)d_in[2];
  const float* bq      = (const float*)d_in[3];
  const float* Wkv     = (const float*)d_in[4];
  const float* bkv     = (const float*)d_in[5];
  const float* Wo      = (const float*)d_in[6];
  const float* bo      = (const float*)d_in[7];
  const float* rpb     = (const float*)d_in[8];
  const int*   rel_idx = (const int*)d_in[9];
  float* out = (float*)d_out;

  swin_local_attn_kernel<<<dim3(NWIN), dim3(256), SMEM_BYTES, stream>>>(
      xa, xb, Wq, bq, Wkv, bkv, Wo, bo, rpb, rel_idx, out);
}
